// Tno_3487513444804
// MI455X (gfx1250) — matrix operand order repaired and checked by reading, not compiled, not run on hardware
//
#include <hip/hip_runtime.h>
#include <cstdint>

// ---------------------------------------------------------------------------
// Types
// ---------------------------------------------------------------------------
typedef __attribute__((ext_vector_type(16))) __bf16 v16bf;
typedef __attribute__((ext_vector_type(8)))  float  v8f;
typedef __attribute__((ext_vector_type(2)))  float  v2f;
typedef __attribute__((ext_vector_type(4)))  unsigned int u32x4;
typedef __attribute__((ext_vector_type(8)))  int    i32x8;
typedef __attribute__((ext_vector_type(4)))  int    i32x4;

#define SEQ      2048
#define NW       2049          // kernel taps 0..2048
#define TPAD     2080          // padded time length (zeros beyond)
#define ODIM     768
#define FDIM     64
#define NBATCH   4

// workspace layout (bytes)
#define H_ROWS      2064                      // 129 tiles of 16
#define H_BYTES     ((size_t)H_ROWS * FDIM * 4)          // 528384
#define WREV_BYTES  ((size_t)ODIM * TPAD * 2)            // 3194880
#define XB_BYTES    ((size_t)ODIM * NBATCH * TPAD * 2)   // 12779520

#if defined(__gfx1250__) && defined(__HIP_DEVICE_COMPILE__) && \
    __has_builtin(__builtin_amdgcn_tensor_load_to_lds) && \
    __has_builtin(__builtin_amdgcn_s_wait_tensorcnt)
#define USE_TDM 1
#endif

// ---------------------------------------------------------------------------
// TDM: 1-D global -> LDS copy, data_size = 8B units (CDNA5 D#, groups 0/1)
// 6-arg builtin variant: (u32x4 g0, i32x8 g1, i32x4 g2, i32x4 g3, i32x8, cpol)
// ---------------------------------------------------------------------------
#if defined(USE_TDM)
__device__ __forceinline__ void tdm_load_1d(unsigned int lds_byte,
                                            const void* gptr,
                                            unsigned int n8 /*8-byte units*/) {
  unsigned long long ga = (unsigned long long)gptr;
  u32x4 g0;
  g0[0] = 1u;                                   // count=1, user mode
  g0[1] = lds_byte;                             // lds_addr
  g0[2] = (unsigned int)ga;                     // global_addr lo
  g0[3] = (unsigned int)((ga >> 32) & 0x1FFFFFFull) | (2u << 30); // hi + type=2
  i32x8 g1;
  g1[0] = (int)(3u << 16);                      // wg_mask=0, data_size=3 (8B)
  g1[1] = (int)((n8 & 0xFFFFu) << 16);          // tensor_dim0 lo16
  g1[2] = (int)((n8 >> 16) & 0xFFFFu) | (1 << 16); // tensor_dim0 hi | tensor_dim1=1
  g1[3] = (int)((n8 & 0xFFFFu) << 16);          // tensor_dim1 hi=0 | tile_dim0=n8
  g1[4] = 1;                                    // tile_dim1=1, tile_dim2=0
  g1[5] = (int)n8;                              // tensor_dim0_stride lo
  g1[6] = 0;                                    // stride0 hi | stride1 lo
  g1[7] = 0;
  i32x4 z4 = {0, 0, 0, 0};
  i32x8 z8 = {0, 0, 0, 0, 0, 0, 0, 0};
  __builtin_amdgcn_tensor_load_to_lds(g0, g1, z4, z4, z8, 0);
}
#endif

// ---------------------------------------------------------------------------
// unaligned 16B gather from LDS via aligned b128/b32 + v_alignbit funnel
// ---------------------------------------------------------------------------
__device__ __forceinline__ void lds_ld16u(const __bf16* base, int byteOff,
                                          unsigned int* out) {
  int a = byteOff & ~15;
  unsigned int sh = (unsigned int)(byteOff & 15) * 8u;
  const char* p = (const char*)base + a;
  uint4 q = *(const uint4*)p;
  unsigned int q4 = *(const unsigned int*)(p + 16);
#if defined(__HIP_DEVICE_COMPILE__) && __has_builtin(__builtin_amdgcn_alignbit)
  out[0] = __builtin_amdgcn_alignbit(q.y, q.x, sh);
  out[1] = __builtin_amdgcn_alignbit(q.z, q.y, sh);
  out[2] = __builtin_amdgcn_alignbit(q.w, q.z, sh);
  out[3] = __builtin_amdgcn_alignbit(q4, q.w, sh);
#else
  out[0] = sh ? ((q.x >> sh) | (q.y << (32 - sh))) : q.x;
  out[1] = sh ? ((q.y >> sh) | (q.z << (32 - sh))) : q.y;
  out[2] = sh ? ((q.z >> sh) | (q.w << (32 - sh))) : q.z;
  out[3] = sh ? ((q.w >> sh) | (q4  << (32 - sh))) : q.w;
#endif
}

// ---------------------------------------------------------------------------
// Kernel 1: x (b,t,d) f32 -> xb[d][b][tpad] bf16, zero padded t>=2048
// ---------------------------------------------------------------------------
__global__ __launch_bounds__(256) void xpose_kernel(const float* __restrict__ x,
                                                    __bf16* __restrict__ xb) {
  int idx = blockIdx.x * 256 + threadIdx.x;      // d*8320 + b*2080 + tp
  if (idx >= ODIM * NBATCH * TPAD) return;
  int tp = idx % TPAD;
  int b  = (idx / TPAD) & 3;
  int d  = idx / (TPAD * NBATCH);
  float v = 0.f;
  if (tp < SEQ) v = x[((size_t)b * SEQ + tp) * ODIM + d];
  xb[idx] = (__bf16)v;
}

// zero the wrev tail pads c = 2048..2079 (c<=2048 written by GEMM kernel)
__global__ __launch_bounds__(256) void wpad_kernel(__bf16* __restrict__ wrev) {
  int idx = blockIdx.x * 256 + threadIdx.x;      // 768 * 32
  if (idx >= ODIM * 32) return;
  wrev[(size_t)(idx >> 5) * TPAD + 2048 + (idx & 31)] = (__bf16)0.f;
}

// ---------------------------------------------------------------------------
// Kernel 2: RPE MLP trunk, wave32-per-row. h: 2 features per lane.
// Outputs H[2064][64] f32 (silu(LN(.)) after 3 mid layers), pad rows zero.
// ---------------------------------------------------------------------------
__device__ __forceinline__ void ln_silu(float h0, float h1,
                                        const float* __restrict__ w,
                                        const float* __restrict__ b,
                                        int lane, float* a0, float* a1) {
  float s = h0 + h1, sq = h0 * h0 + h1 * h1;
  for (int off = 16; off > 0; off >>= 1) {
    s  += __shfl_xor(s,  off, 32);
    sq += __shfl_xor(sq, off, 32);
  }
  float mean = s * (1.f / 64.f);
  float var  = sq * (1.f / 64.f) - mean * mean;
  float inv  = rsqrtf(var + 1e-5f);
  float g0 = (h0 - mean) * inv * w[lane]      + b[lane];
  float g1 = (h1 - mean) * inv * w[lane + 32] + b[lane + 32];
  *a0 = g0 / (1.f + __expf(-g0));
  *a1 = g1 / (1.f + __expf(-g1));
}

__global__ __launch_bounds__(256) void trunk_kernel(
    const float* __restrict__ w_in, const float* __restrict__ ln_w,
    const float* __restrict__ ln_b, const float* __restrict__ w_mid,
    const float* __restrict__ ln_out_w, const float* __restrict__ ln_out_b,
    float* __restrict__ H) {
  __shared__ float Wt[3 * FDIM * FDIM];          // transposed: Wt[l][k*64+f]
  const int tid = threadIdx.x;
  for (int i = tid; i < 3 * FDIM * FDIM; i += 256) {
    int l = i >> 12, rem = i & 4095, f = rem >> 6, k = rem & 63;
    Wt[(l << 12) + (k << 6) + f] = w_mid[i];     // w_mid[l][f][k]
  }
  __syncthreads();
  const int w = tid >> 5, lane = tid & 31;
  const float wi0 = w_in[lane], wi1 = w_in[lane + 32];
  for (int row = blockIdx.x * 8 + w; row < H_ROWS; row += gridDim.x * 8) {
    if (row > SEQ) {                              // zero pad rows
      H[(size_t)row * FDIM + lane] = 0.f;
      H[(size_t)row * FDIM + lane + 32] = 0.f;
      continue;
    }
    float t = (float)row;
    float h0 = t * wi0, h1 = t * wi1;
    for (int l = 0; l < 3; ++l) {
      float a0, a1;
      ln_silu(h0, h1, ln_w + l * FDIM, ln_b + l * FDIM, lane, &a0, &a1);
      const float* W = Wt + (l << 12);
      float n0 = 0.f, n1 = 0.f;
      #pragma unroll 8
      for (int k = 0; k < 32; ++k) {
        float ak = __shfl(a0, k, 32);
        n0 = fmaf(ak, W[(k << 6) + lane], n0);
        n1 = fmaf(ak, W[(k << 6) + lane + 32], n1);
      }
      #pragma unroll 8
      for (int k = 0; k < 32; ++k) {
        float ak = __shfl(a1, k, 32);
        n0 = fmaf(ak, W[((k + 32) << 6) + lane], n0);
        n1 = fmaf(ak, W[((k + 32) << 6) + lane + 32], n1);
      }
      h0 = n0; h1 = n1;
    }
    float a0, a1;
    ln_silu(h0, h1, ln_out_w, ln_out_b, lane, &a0, &a1);
    H[(size_t)row * FDIM + lane] = a0;
    H[(size_t)row * FDIM + lane + 32] = a1;
  }
}

// ---------------------------------------------------------------------------
// Kernel 3: W = H @ w_out^T via v_wmma_f32_16x16x4_f32, apply decay,
// store reversed bf16: wrev[j][2048 - t] = W[t][j] * sigmoid(gamma_j)^t
// wave-per-tile: M=16 rows(t), N=16 cols(j), K=64 in steps of 4
// ---------------------------------------------------------------------------
__global__ __launch_bounds__(256) void rpe_out_kernel(
    const float* __restrict__ H, const float* __restrict__ w_out,
    const float* __restrict__ gamma, __bf16* __restrict__ wrev) {
#if defined(__HIP_DEVICE_COMPILE__)
  const int tid = threadIdx.x, w = tid >> 5, lane = tid & 31;
  const int tileT = blockIdx.y * 8 + w;
  if (tileT >= (H_ROWS / 16)) return;            // 129 tiles
  const int T0 = tileT * 16, J0 = blockIdx.x * 16;
  const int half = lane >> 4, n = lane & 15;
  const int j = J0 + n;
  // A: 16x4 f32: row M = lane&15, K = K0 + half*2 + {0,1}
  const float* hA = H     + (size_t)(T0 + n) * FDIM + half * 2;
  // B: 4x16 f32: col N = lane&15, K = K0 + half*2 + {0,1}  (B[k][n] = w_out[j][k])
  const float* bB = w_out + (size_t)j * FDIM + half * 2;
  v8f acc = {0.f, 0.f, 0.f, 0.f, 0.f, 0.f, 0.f, 0.f};
  #pragma unroll
  for (int K0 = 0; K0 < FDIM; K0 += 4) {
    v2f a; a.x = hA[K0]; a.y = hA[K0 + 1];
    v2f b; b.x = bB[K0]; b.y = bB[K0 + 1];
    acc = __builtin_amdgcn_wmma_f32_16x16x4_f32(false, a, false, b,
                                                (short)0, acc, false, false);
  }
  float g = gamma[j];
  float lsg = fminf(g, 0.f) - log1pf(__expf(-fabsf(g)));  // logsigmoid(g)
  #pragma unroll
  for (int r = 0; r < 8; ++r) {
    int t = T0 + half * 8 + r;                   // C: VGPR r -> M = half*8 + r
    if (t <= SEQ) {
      float val = acc[r] * __expf((float)t * lsg);
      wrev[(size_t)j * TPAD + (SEQ - t)] = (__bf16)val;
    }
  }
#endif
}

// ---------------------------------------------------------------------------
// Kernel 4: causal conv as chunked Toeplitz bf16 WMMA.
// One workgroup per channel d; TDM-stage wrev[d] (4.2KB) + xb[d] (16.6KB) to
// LDS; 8 waves x 16 t-tiles; K-loop over 32-lag chunks.
//   A[m,k] = w[(T0-U0)+m-k] = wrev[2048-(T0-U0)-m+k]  (contiguous in k)
//   B[k,n] = xb[d][n][U0+k]  (n = batch, cols 4..15 are don't-care)
// ---------------------------------------------------------------------------
__global__ __launch_bounds__(256) void conv_kernel(const __bf16* __restrict__ xb,
                                                   const __bf16* __restrict__ wrev,
                                                   float* __restrict__ y) {
#if defined(__HIP_DEVICE_COMPILE__)
  __shared__ __align__(16) __bf16 xb_s[NBATCH * TPAD];   // 16640 B
  __shared__ __align__(16) __bf16 wrev_s[TPAD + 8];      // 4160 B + slack
  const int d = blockIdx.x;
  const int tid = threadIdx.x;
#if defined(USE_TDM)
  if (tid == 0) {
    tdm_load_1d((unsigned int)(size_t)(void*)xb_s,
                xb + (size_t)d * NBATCH * TPAD, (NBATCH * TPAD * 2) / 8);
    tdm_load_1d((unsigned int)(size_t)(void*)wrev_s,
                wrev + (size_t)d * TPAD, (TPAD * 2) / 8);
    __builtin_amdgcn_s_wait_tensorcnt(0);
  }
#else
  {
    const unsigned int* gx = (const unsigned int*)(xb + (size_t)d * NBATCH * TPAD);
    unsigned int* sx = (unsigned int*)xb_s;
    for (int i = tid; i < (NBATCH * TPAD) / 2; i += 256) sx[i] = gx[i];
    const unsigned int* gw = (const unsigned int*)(wrev + (size_t)d * TPAD);
    unsigned int* sw = (unsigned int*)wrev_s;
    for (int i = tid; i < TPAD / 2; i += 256) sw[i] = gw[i];
    if (tid < 4) sw[TPAD / 2 + tid] = 0u;        // slack
  }
#endif
  __syncthreads();

  const int w = tid >> 5, lane = tid & 31;
  const int half = lane >> 4, m = lane & 15, n = lane & 15;
  const int bclamp = lane & 3;                   // cols 4..15 read dup data
  const int bBase = (bclamp * TPAD + half * 16) * 2;

  for (int tile = w; tile < SEQ / 16; tile += 8) {
    const int T0 = tile << 4;
    v8f acc = {0.f, 0.f, 0.f, 0.f, 0.f, 0.f, 0.f, 0.f};
    // A lane k-window start: k0 = half*16 (16 contiguous k)
    int aOff = (2048 - T0 - m + half * 16) * 2;  // bytes into wrev_s at U0=0
    int bOff = bBase;                            // bytes into xb_s at U0=0
    for (int U0 = 0; U0 <= T0; U0 += 32, aOff += 64, bOff += 64) {
      union { unsigned int u[8]; v16bf v; } A;
      lds_ld16u(wrev_s, aOff,      &A.u[0]);     // k = k0 .. k0+7
      lds_ld16u(wrev_s, aOff + 16, &A.u[4]);     // k = k0+8 .. k0+15
      union { unsigned int u[8]; v16bf v; } B;
      const uint4* bp = (const uint4*)((const char*)xb_s + bOff);
      *(uint4*)&B.u[0] = bp[0];                  // k = half*16 .. +7
      *(uint4*)&B.u[4] = bp[1];                  // k = half*16+8 .. +15
      acc = __builtin_amdgcn_wmma_f32_16x16x32_bf16(false, A.v, false, B.v,
                                                    (short)0, acc, false, false);
    }
    if (n < NBATCH) {
      float* yp = y + ((size_t)n * SEQ + T0 + half * 8) * ODIM + d;
      #pragma unroll
      for (int r = 0; r < 8; ++r) yp[(size_t)r * ODIM] = acc[r];
    }
  }
#endif
}

// ---------------------------------------------------------------------------
// launcher
// ---------------------------------------------------------------------------
extern "C" void kernel_launch(void* const* d_in, const int* in_sizes, int n_in,
                              void* d_out, int out_size, void* d_ws, size_t ws_size,
                              hipStream_t stream) {
  const float* x        = (const float*)d_in[0];
  const float* w_in     = (const float*)d_in[1];
  const float* ln_w     = (const float*)d_in[2];
  const float* ln_b     = (const float*)d_in[3];
  const float* w_mid    = (const float*)d_in[4];
  const float* ln_out_w = (const float*)d_in[5];
  const float* ln_out_b = (const float*)d_in[6];
  const float* w_out    = (const float*)d_in[7];
  const float* gamma    = (const float*)d_in[8];
  float* y = (float*)d_out;

  char* ws = (char*)d_ws;
  float*  H    = (float*)ws;
  __bf16* wrev = (__bf16*)(ws + H_BYTES);
  __bf16* xb   = (__bf16*)(ws + H_BYTES + WREV_BYTES);

  xpose_kernel<<<(ODIM * NBATCH * TPAD + 255) / 256, 256, 0, stream>>>(x, xb);
  wpad_kernel<<<(ODIM * 32 + 255) / 256, 256, 0, stream>>>(wrev);
  trunk_kernel<<<32, 256, 0, stream>>>(w_in, ln_w, ln_b, w_mid,
                                       ln_out_w, ln_out_b, H);
  rpe_out_kernel<<<dim3(ODIM / 16, 17), 256, 0, stream>>>(H, w_out, gamma, wrev);
  conv_kernel<<<ODIM, 256, 0, stream>>>(xb, wrev, y);
}
